// NonLocal_21646635171913
// MI455X (gfx1250) — matrix operand order repaired and checked by reading, not compiled, not run on hardware
//
#include <hip/hip_runtime.h>
#include <hip/hip_bf16.h>

// ---------------------------------------------------------------------------
// NonLocal block on gfx1250 (MI455X): all GEMMs via v_wmma_f32_16x16x32_bf16.
// Shapes: B=8, C=512, H=W=64, IC=256, N(view)=2048, pixels=4096.
// Block = 256 threads = 8 waves; block tile 128x128; wave tile 32x64
// (2 A-frags x 4 B-frags = 8 WMMAs per K=32 step).
// ---------------------------------------------------------------------------

#define USE_ASYNC_LDS 1

typedef __attribute__((ext_vector_type(8)))  float  v8f;
typedef __attribute__((ext_vector_type(16))) __bf16 v16bf;
typedef __attribute__((ext_vector_type(8)))  __bf16 v8bf;

union Frag16 {
    v16bf v;
    struct { v8bf lo; v8bf hi; } p;
};

__device__ __forceinline__ unsigned short f32_to_bf16_raw(float f) {
    union { float f; unsigned u; } x; x.f = f;
    unsigned r = x.u + 0x7FFFu + ((x.u >> 16) & 1u);   // round-to-nearest-even
    return (unsigned short)(r >> 16);
}

// One K=32 step for one wave of an 8-wave block (128x128 tile).
// Wave (wr, wc) computes rows [wr*32, wr*32+32) x cols [wc*64, wc*64+64).
// lA : [128 rows(M)][40] bf16 bits, row r holds K=0..31 contiguously.
// lBt: [128 cols(N)][40] bf16 bits, col n holds K=0..31 contiguously.
// A frag (ISA 16x32): lane half h, elem i<8 -> K=h*8+i, i>=8 -> K=16+h*8+(i-8).
// B frag (ISA 32x16): lane half h, elem i<8 -> K=h*8+i, i>=8 -> K=16+h*8+(i-8).
__device__ __forceinline__ void wave_mma_step(const unsigned short (*lA)[40],
                                              const unsigned short (*lBt)[40],
                                              int wr, int wc, int half, int idx,
                                              v8f acc[2][4]) {
    Frag16 a[2];
#pragma unroll
    for (int u = 0; u < 2; ++u) {
        const unsigned short* ar = lA[wr * 32 + u * 16 + idx];
        a[u].p.lo = *(const v8bf*)(ar + half * 8);
        a[u].p.hi = *(const v8bf*)(ar + 16 + half * 8);
    }
#pragma unroll
    for (int s = 0; s < 4; ++s) {
        Frag16 b;
        const unsigned short* br = lBt[wc * 64 + s * 16 + idx];
        b.p.lo = *(const v8bf*)(br + half * 8);
        b.p.hi = *(const v8bf*)(br + 16 + half * 8);
#pragma unroll
        for (int u = 0; u < 2; ++u)
            acc[u][s] = __builtin_amdgcn_wmma_f32_16x16x32_bf16(
                false, a[u].v, false, b.v, (short)0, acc[u][s], false, false);
    }
}

// ---------------------------------------------------------------------------
// K1: projection GEMM  Out[256,4096] = W[256,512] @ Src[512,4096] + bias
// transposed==0 : natural row-major [256,4096] (== [512,2048] view)  (phi)
// transposed==1 : transposed VIEW [2048,512]:
//                 element (oc,pix) -> T[(pix&2047)*512 + 2*oc + (pix>>11)]   (theta, g)
// ---------------------------------------------------------------------------
__global__ void __launch_bounds__(256)
proj_kernel(const float* __restrict__ W, const float* __restrict__ bias,
            const float* __restrict__ Src, unsigned short* __restrict__ Out,
            int transposed) {
    const int b      = blockIdx.z;
    const int tileN0 = blockIdx.x * 128;   // pixel
    const int tileM0 = blockIdx.y * 128;   // oc
    const float* src = Src + (size_t)b * 512 * 4096;
    unsigned short* out = Out + (size_t)b * 512 * 2048;

    __shared__ unsigned short lA[128][40];
    __shared__ unsigned short lBt[128][40];

    const int t = threadIdx.x;
    const int wv = t >> 5, lane = t & 31, half = lane >> 4, idx = lane & 15;
    const int wr = wv >> 1, wc = wv & 1;
    v8f acc[2][4] = {};

    for (int k0 = 0; k0 < 512; k0 += 32) {
        __syncthreads();
        for (int i = t; i < 128 * 32; i += 256) {          // A: weights (f32->bf16)
            int r = i >> 5, c = i & 31;
            lA[r][c] = f32_to_bf16_raw(W[(size_t)(tileM0 + r) * 512 + k0 + c]);
        }
        for (int i = t; i < 32 * 128; i += 256) {          // B: activations (transposed)
            int k = i >> 7, n = i & 127;
            lBt[n][k] = f32_to_bf16_raw(src[(size_t)(k0 + k) * 4096 + tileN0 + n]);
        }
        __syncthreads();
        wave_mma_step(lA, lBt, wr, wc, half, idx, acc);
    }
#pragma unroll
    for (int s = 0; s < 4; ++s) {
        int pix = tileN0 + wc * 64 + s * 16 + idx;
#pragma unroll
        for (int u = 0; u < 2; ++u) {
#pragma unroll
            for (int j = 0; j < 8; ++j) {
                int oc = tileM0 + wr * 32 + u * 16 + half * 8 + j;
                unsigned short h = f32_to_bf16_raw(acc[u][s][j] + bias[oc]);
                if (!transposed)
                    out[(size_t)oc * 4096 + pix] = h;
                else
                    out[(size_t)(pix & 2047) * 512 + 2 * oc + (pix >> 11)] = h;
            }
        }
    }
}

// ---------------------------------------------------------------------------
// K2: scores[n,m] = sum_c thetaT[n,c] * phi_view[c,m]    (2048x2048, K=512)
//     A tile is a raw bf16 copy -> async DMA to LDS (ASYNCcnt path).
// ---------------------------------------------------------------------------
__global__ void __launch_bounds__(256)
scores_kernel(const unsigned short* __restrict__ ThetaT,  // [B,2048,512]
              const unsigned short* __restrict__ Phi,     // [B,512,2048]
              float* __restrict__ S) {                    // [B,2048,2048]
    const int b      = blockIdx.z;
    const int tileN0 = blockIdx.x * 128;   // m
    const int tileM0 = blockIdx.y * 128;   // n
    const unsigned short* At = ThetaT + (size_t)b * 2048 * 512;
    const unsigned short* Bp = Phi    + (size_t)b * 512 * 2048;
    float* Sb = S + (size_t)b * 2048 * 2048;

    __shared__ unsigned short lA[128][40];
    __shared__ unsigned short lBt[128][40];

    const int t = threadIdx.x;
    const int wv = t >> 5, lane = t & 31, half = lane >> 4, idx = lane & 15;
    const int wr = wv >> 1, wc = wv & 1;
    v8f acc[2][4] = {};

    for (int k0 = 0; k0 < 512; k0 += 32) {
        __syncthreads();
#if USE_ASYNC_LDS
        // A tile: 128 rows x 64B, copied 16B per lane -> 2 async b128 per thread.
#pragma unroll
        for (int it = 0; it < 2; ++it) {
            int e = t + it * 256;          // 512 x 16B chunks
            int r = e >> 2, c8 = (e & 3) * 8;
            unsigned ldsa = (unsigned)(size_t)&lA[r][c8];
            const unsigned short* g = At + (size_t)(tileM0 + r) * 512 + k0 + c8;
            asm volatile("global_load_async_to_lds_b128 %0, %1, off"
                         :: "v"(ldsa), "v"(g) : "memory");
        }
#else
        for (int i = t; i < 128 * 32; i += 256) {
            int r = i >> 5, c = i & 31;
            lA[r][c] = At[(size_t)(tileM0 + r) * 512 + k0 + c];
        }
#endif
        for (int i = t; i < 32 * 128; i += 256) {          // B (transposed into LDS)
            int k = i >> 7, n = i & 127;
            lBt[n][k] = Bp[(size_t)(k0 + k) * 2048 + tileN0 + n];
        }
#if USE_ASYNC_LDS
        asm volatile("s_wait_asynccnt 0x0" ::: "memory");
#endif
        __syncthreads();
        wave_mma_step(lA, lBt, wr, wc, half, idx, acc);
    }
#pragma unroll
    for (int s = 0; s < 4; ++s) {
        int m = tileN0 + wc * 64 + s * 16 + idx;
#pragma unroll
        for (int u = 0; u < 2; ++u) {
#pragma unroll
            for (int j = 0; j < 8; ++j) {
                int n = tileM0 + wr * 32 + u * 16 + half * 8 + j;
                Sb[(size_t)n * 2048 + m] = acc[u][s][j];
            }
        }
    }
}

// ---------------------------------------------------------------------------
// K3: column softmax stats (softmax over axis n, per column m): online max/sum
// ---------------------------------------------------------------------------
__global__ void __launch_bounds__(256)
colstats_kernel(const float* __restrict__ S,
                float* __restrict__ cmax, float* __restrict__ csum) {
    const int b = blockIdx.y;
    const int m = blockIdx.x * 256 + threadIdx.x;
    const float* Sb = S + (size_t)b * 2048 * 2048 + m;
    float mx = -3.4e38f, sum = 0.f;
    for (int n = 0; n < 2048; ++n) {
        float s  = Sb[(size_t)n * 2048];
        float nm = fmaxf(mx, s);
        sum = sum * __expf(mx - nm) + __expf(s - nm);
        mx  = nm;
    }
    cmax[b * 2048 + m] = mx;
    csum[b * 2048 + m] = sum;
}

// ---------------------------------------------------------------------------
// K4: out_nl[n,c] = sum_m softmax(S)[n,m] * gT[m,c]   (2048x512, K=2048)
//     attn materialized on the fly while filling the A tile.
// ---------------------------------------------------------------------------
__global__ void __launch_bounds__(256)
outnl_kernel(const float* __restrict__ S,
             const float* __restrict__ cmax, const float* __restrict__ csum,
             const unsigned short* __restrict__ GT,     // [B,2048,512]
             unsigned short* __restrict__ Onl) {        // [B,2048,512]
    const int b      = blockIdx.z;
    const int tileN0 = blockIdx.x * 128;   // c
    const int tileM0 = blockIdx.y * 128;   // n
    const float* Sb = S + (size_t)b * 2048 * 2048;
    const float* cm = cmax + b * 2048;
    const float* cs = csum + b * 2048;
    const unsigned short* G = GT + (size_t)b * 2048 * 512;
    unsigned short* Ob = Onl + (size_t)b * 2048 * 512;

    __shared__ unsigned short lA[128][40];
    __shared__ unsigned short lBt[128][40];

    const int t = threadIdx.x;
    const int wv = t >> 5, lane = t & 31, half = lane >> 4, idx = lane & 15;
    const int wr = wv >> 1, wc = wv & 1;
    v8f acc[2][4] = {};

    for (int k0 = 0; k0 < 2048; k0 += 32) {
        __syncthreads();
        for (int i = t; i < 128 * 32; i += 256) {          // A: attn tile
            int r = i >> 5, c = i & 31;
            int m = k0 + c;
            float s = Sb[(size_t)(tileM0 + r) * 2048 + m];
            float a = __expf(s - cm[m]) / cs[m];
            lA[r][c] = f32_to_bf16_raw(a);
        }
        for (int i = t; i < 32 * 128; i += 256) {          // B: gT tile (transposed)
            int k = i >> 7, n = i & 127;
            lBt[n][k] = G[(size_t)(k0 + k) * 512 + tileN0 + n];
        }
        __syncthreads();
        wave_mma_step(lA, lBt, wr, wc, half, idx, acc);
    }
#pragma unroll
    for (int s = 0; s < 4; ++s) {
        int c = tileN0 + wc * 64 + s * 16 + idx;
#pragma unroll
        for (int u = 0; u < 2; ++u) {
#pragma unroll
            for (int j = 0; j < 8; ++j) {
                int n = tileM0 + wr * 32 + u * 16 + half * 8 + j;
                Ob[(size_t)n * 512 + c] = f32_to_bf16_raw(acc[u][s][j]);
            }
        }
    }
}

// ---------------------------------------------------------------------------
// K5: out = w_mask[512,256] @ out_rs[256,4096] + b_mask + x
//     out_rs[ic,pix] = out_nl[pix&2047, 2*ic + (pix>>11)]  (view un-reshape)
// ---------------------------------------------------------------------------
__global__ void __launch_bounds__(256)
final_kernel(const float* __restrict__ Wm, const float* __restrict__ bm,
             const unsigned short* __restrict__ Onl,    // [B,2048,512]
             const float* __restrict__ X,
             float* __restrict__ Out) {
    const int b      = blockIdx.z;
    const int tileN0 = blockIdx.x * 128;   // pix
    const int tileM0 = blockIdx.y * 128;   // o
    const unsigned short* O = Onl + (size_t)b * 2048 * 512;

    __shared__ unsigned short lA[128][40];
    __shared__ unsigned short lBt[128][40];

    const int t = threadIdx.x;
    const int wv = t >> 5, lane = t & 31, half = lane >> 4, idx = lane & 15;
    const int wr = wv >> 1, wc = wv & 1;
    v8f acc[2][4] = {};

    for (int k0 = 0; k0 < 256; k0 += 32) {
        __syncthreads();
        for (int i = t; i < 128 * 32; i += 256) {          // A: w_mask (f32->bf16)
            int r = i >> 5, c = i & 31;
            lA[r][c] = f32_to_bf16_raw(Wm[(size_t)(tileM0 + r) * 256 + k0 + c]);
        }
        for (int i = t; i < 32 * 128; i += 256) {          // B: out_nl via view
            int k = i & 31, n = i >> 5;                    // consecutive t -> consecutive ic
            int pix = tileN0 + n;
            int ic  = k0 + k;
            lBt[n][k] = O[(size_t)(pix & 2047) * 512 + 2 * ic + (pix >> 11)];
        }
        __syncthreads();
        wave_mma_step(lA, lBt, wr, wc, half, idx, acc);
    }
#pragma unroll
    for (int s = 0; s < 4; ++s) {
        int pix = tileN0 + wc * 64 + s * 16 + idx;
#pragma unroll
        for (int u = 0; u < 2; ++u) {
#pragma unroll
            for (int j = 0; j < 8; ++j) {
                int o = tileM0 + wr * 32 + u * 16 + half * 8 + j;
                size_t a = ((size_t)b * 512 + o) * 4096 + pix;
                Out[a] = acc[u][s][j] + bm[o] + X[a];
            }
        }
    }
}

// ---------------------------------------------------------------------------
extern "C" void kernel_launch(void* const* d_in, const int* in_sizes, int n_in,
                              void* d_out, int out_size, void* d_ws, size_t ws_size,
                              hipStream_t stream) {
    const float* x       = (const float*)d_in[0];
    const float* y       = (const float*)d_in[1];
    const float* w_phi   = (const float*)d_in[2];
    const float* b_phi   = (const float*)d_in[3];
    const float* w_theta = (const float*)d_in[4];
    const float* b_theta = (const float*)d_in[5];
    const float* w_g     = (const float*)d_in[6];
    const float* b_g     = (const float*)d_in[7];
    const float* w_mask  = (const float*)d_in[8];
    const float* b_mask  = (const float*)d_in[9];
    float* out = (float*)d_out;
    char* ws = (char*)d_ws;

    const size_t actB  = 8ull * 512 * 2048 * sizeof(unsigned short);   // 16 MiB each
    size_t off = 0;
    unsigned short* phi    = (unsigned short*)(ws + off); off += actB; // [B,512,2048] view
    unsigned short* thetaT = (unsigned short*)(ws + off); off += actB; // [B,2048,512]
    unsigned short* gT     = (unsigned short*)(ws + off); off += actB; // [B,2048,512]
    float* S    = (float*)(ws + off); off += 8ull * 2048 * 2048 * sizeof(float); // 128 MiB
    float* cmax = (float*)(ws + off); off += 8ull * 2048 * sizeof(float);
    float* csum = (float*)(ws + off); off += 8ull * 2048 * sizeof(float);
    unsigned short* onl = (unsigned short*)(ws + off); off += actB;    // [B,2048,512]

    dim3 blk(256);
    proj_kernel<<<dim3(32, 2, 8), blk, 0, stream>>>(w_phi,   b_phi,   y, phi,    0);
    proj_kernel<<<dim3(32, 2, 8), blk, 0, stream>>>(w_theta, b_theta, x, thetaT, 1);
    proj_kernel<<<dim3(32, 2, 8), blk, 0, stream>>>(w_g,     b_g,     x, gT,     1);
    scores_kernel<<<dim3(16, 16, 8), blk, 0, stream>>>(thetaT, phi, S);
    colstats_kernel<<<dim3(8, 8), 256, 0, stream>>>(S, cmax, csum);
    outnl_kernel<<<dim3(4, 16, 8), blk, 0, stream>>>(S, cmax, csum, gT, onl);
    final_kernel<<<dim3(32, 4, 8), blk, 0, stream>>>(w_mask, b_mask, onl, x, out);
}
